// LSTM_Cell_38491496907449
// MI455X (gfx1250) — matrix operand order repaired and checked by reading, not compiled, not run on hardware
//
#include <hip/hip_runtime.h>
#include <stdint.h>

// ---------------------------------------------------------------------------
// Fused LSTM cell for MI455X (gfx1250, wave32, WMMA).
// Pass 1: pack f32 -> bf16 once (A = [x | a_prev] as [8192,2048],
//         B = [Wg | Ug] per gate as [4,1024,2048], bias = b+bu).
// Pass 2: bf16 WMMA GEMM (f32 accum) fused with bias + sigmoid/tanh gating.
//         A tiles double-buffered in LDS via gfx1250 async-to-LDS (ASYNCcnt
//         wait pipelined one iteration behind the issue). Waves arranged
//         2(M)x4(N): each B fragment feeds 4 WMMAs, halving global B-load
//         issue vs a 4x2 arrangement. K-loop at unroll 1 + peeled tail keeps
//         the kernel under 256 VGPRs (no vgpr_msb churn).
// ---------------------------------------------------------------------------

typedef __bf16 bf16_t;
typedef bf16_t v16bf __attribute__((ext_vector_type(16)));
typedef float  v8f   __attribute__((ext_vector_type(8)));

#define LSTM_BATCH 8192
#define LSTM_HID   1024
#define LSTM_KIN   1024
#define LSTM_KTOT  2048   // x-K (1024) followed by a_prev-K (1024)
#define TILE_BM    128
#define TILE_BN    64
#define TILE_KT    32
#define K_ITERS    (LSTM_KTOT / TILE_KT)   // 64

struct LstmParams {
  const float* __restrict__ x;     // [8192,1024]
  const float* __restrict__ a;     // [8192,1024]
  const float* __restrict__ c;     // [8192,1024]
  const float* __restrict__ W[4];  // f,i,o,c : [1024,1024] row-major [out,in]
  const float* __restrict__ U[4];
  const float* __restrict__ b[4];  // [1024]
  const float* __restrict__ bu[4];
  float* __restrict__ out_a;       // [8192,1024]
  float* __restrict__ out_c;       // [8192,1024]
  bf16_t* __restrict__ A_bf;       // ws: [8192][2048]
  bf16_t* __restrict__ B_bf;       // ws: [4][1024][2048]
  float*  __restrict__ bsum;       // ws: [4][1024]
};

union BF8  { bf16_t h[8];  uint4 q;    };
union Frag { v16bf  v;     uint4 q[2]; };

__device__ __forceinline__ uint4 cvt8_f32_bf16(float4 a, float4 b) {
  BF8 u;
  u.h[0] = (bf16_t)a.x; u.h[1] = (bf16_t)a.y;
  u.h[2] = (bf16_t)a.z; u.h[3] = (bf16_t)a.w;
  u.h[4] = (bf16_t)b.x; u.h[5] = (bf16_t)b.y;
  u.h[6] = (bf16_t)b.z; u.h[7] = (bf16_t)b.w;
  return u.q;
}

__device__ __forceinline__ float sigmoid_f(float x) {
  return 1.0f / (1.0f + __expf(-x));
}

// Issue this thread's 32B slice of one A tile as two async b128 copies.
__device__ __forceinline__ void async_fill(unsigned lds, const bf16_t* gsrc) {
  const uint64_t ga = (uint64_t)(const void*)gsrc;
  asm volatile(
      "global_load_async_to_lds_b128 %0, %1, off\n\t"
      "global_load_async_to_lds_b128 %2, %3, off offset:16"
      :
      : "v"(lds), "v"(ga), "v"(lds + 16u), "v"(ga + 16u)
      : "memory");
}

// One K-step: 4 A fragments from LDS, 4 B fragments from global (one per
// gate, each feeding 4 WMMAs), 16 WMMAs total.
__device__ __forceinline__ void mma_step(
    const bf16_t* __restrict__ Bbf, const bf16_t* sAcur,
    int k, int arow, int akc, int ncol, int bkc,
    v8f (&acc)[4][4]) {
  Frag aF[4];
#pragma unroll
  for (int mt = 0; mt < 4; ++mt) {
    const bf16_t* sp = sAcur + (arow + mt * 16) * TILE_KT;
    aF[mt].q[0] = *(const uint4*)(sp + akc);
    aF[mt].q[1] = *(const uint4*)(sp + akc + 16);
  }
#pragma unroll
  for (int g = 0; g < 4; ++g) {
    const bf16_t* bp = Bbf +
        (size_t)(g * LSTM_HID + ncol) * LSTM_KTOT + k + bkc;
    Frag bF;
    bF.q[0] = ((const uint4*)bp)[0];
    bF.q[1] = ((const uint4*)bp)[2];
#pragma unroll
    for (int mt = 0; mt < 4; ++mt) {
      acc[g][mt] = __builtin_amdgcn_wmma_f32_16x16x32_bf16(
          /*neg_a=*/false, aF[mt].v,
          /*neg_b=*/false, bF.v,
          /*c_mod=*/(short)0, acc[g][mt],
          /*reuse_a=*/false, /*reuse_b=*/false);
    }
  }
}

// ---- Pass 1a: pack [x | a_prev] -> bf16 A_cat [8192][2048] ----------------
__global__ __launch_bounds__(256) void pack_a_kernel(LstmParams P) {
  const size_t e = ((size_t)blockIdx.x * 256 + threadIdx.x) * 8;  // elem index
  const int    bm   = (int)(e >> 11);         // batch row
  const int    kcol = (int)(e & 2047);        // 0..2047, multiple of 8
  const float* src  = (kcol < LSTM_KIN)
                        ? (P.x + (size_t)bm * LSTM_KIN + kcol)
                        : (P.a + (size_t)bm * LSTM_KIN + (kcol - LSTM_KIN));
  float4 f0 = ((const float4*)src)[0];
  float4 f1 = ((const float4*)src)[1];
  *(uint4*)(P.A_bf + e) = cvt8_f32_bf16(f0, f1);
}

// ---- Pass 1b: pack [Wg | Ug] -> bf16 B_cat [4][1024][2048] ----------------
__global__ __launch_bounds__(256) void pack_b_kernel(LstmParams P) {
  const size_t e = ((size_t)blockIdx.x * 256 + threadIdx.x) * 8;
  const int    g    = (int)(e >> 21);         // gate
  const size_t rem  = e & ((1u << 21) - 1);
  const int    n    = (int)(rem >> 11);       // output column (hidden idx)
  const int    kcol = (int)(rem & 2047);
  const float* src  = (kcol < LSTM_KIN)
                        ? (P.W[g] + (size_t)n * LSTM_KIN + kcol)
                        : (P.U[g] + (size_t)n * LSTM_KIN + (kcol - LSTM_KIN));
  float4 f0 = ((const float4*)src)[0];
  float4 f1 = ((const float4*)src)[1];
  *(uint4*)(P.B_bf + e) = cvt8_f32_bf16(f0, f1);
}

// ---- Pass 1c: bsum[g][h] = b_g[h] + bu_g[h] -------------------------------
__global__ __launch_bounds__(256) void pack_bias_kernel(LstmParams P) {
  const int i = blockIdx.x * 256 + threadIdx.x;   // 0..4095
  const int g = i >> 10;
  const int h = i & 1023;
  P.bsum[i] = P.b[g][h] + P.bu[g][h];
}

// ---- Pass 2: fused WMMA GEMM + LSTM gating --------------------------------
__launch_bounds__(256)
__global__ void lstm_wmma_kernel(LstmParams P) {
  // Double-buffered A tile: 2 x (128 rows x 32 K) bf16 = 16KB.
  __shared__ __align__(16) bf16_t sA[2][TILE_BM][TILE_KT];

  const int tid    = threadIdx.x;
  const int lane   = tid & 31;
  const int wid    = tid >> 5;
  const int wm     = wid & 1;   // wave M position: 64 rows each (2 waves)
  const int wn     = wid >> 1;  // wave N position: 16 cols each (4 waves)
  const int blockM = blockIdx.y * TILE_BM;
  const int blockN = blockIdx.x * TILE_BN;

  // Accumulators: 4 gates x 4 M-tiles of 16x16 f32 (128 VGPRs).
  v8f acc[4][4];
#pragma unroll
  for (int g = 0; g < 4; ++g)
#pragma unroll
    for (int mt = 0; mt < 4; ++mt)
#pragma unroll
      for (int i = 0; i < 8; ++i)
        acc[g][mt][i] = 0.0f;

  // Async staging map: each of 256 threads moves 32B (two b128 chunks):
  // row = tid>>1, halves [ (tid&1)*16 , +16 ).
  const int sr  = tid >> 1;
  const int sc0 = (tid & 1) * 16;
  const bf16_t*  gA    = P.A_bf + (size_t)(blockM + sr) * LSTM_KTOT + sc0;
  const unsigned ldsA0 = (unsigned)(size_t)&sA[0][sr][sc0];  // LDS byte offset
  const unsigned ldsA1 = (unsigned)(size_t)&sA[1][sr][sc0];

  // A-fragment addressing (ISA 16-bit A layout):
  //   lanes 0-15 : row M = lane,    K chunks [0..7]  and [16..23]
  //   lanes 16-31: row M = lane-16, K chunks [8..15] and [24..31]
  const int arow = wm * 64 + (lane & 15);
  const int akc  = (lane < 16) ? 0 : 8;

  // B-fragment addressing: column N = lane&15; lanes 0-15 hold K [0..7] and
  // [16..23], lanes 16-31 hold K [8..15] and [24..31] (same as A).
  const int ncol = blockN + wn * 16 + (lane & 15);
  const int bkc  = (lane < 16) ? 0 : 8;

  // Prologue: start filling buffer 0 with the first K-panel.
  async_fill(ldsA0, gA);

  // Steady state: always prefetch the next panel, wait only for the
  // one-iteration-old copies (in-order async completion => <=2 outstanding
  // means the current buffer has landed). unroll 1 keeps one iteration of
  // B fragments in flight and the kernel under 256 VGPRs.
#pragma unroll 1
  for (int it = 0; it < K_ITERS - 1; ++it) {
    const int cur = it & 1;
    __syncthreads();  // WAR: everyone done reading the buffer we overwrite
    async_fill(cur ? ldsA0 : ldsA1, gA + (it + 1) * TILE_KT);
    asm volatile("s_wait_asynccnt 0x2" ::: "memory");
    __syncthreads();  // current buffer fully populated by all threads
    mma_step(P.B_bf, &sA[cur][0][0], it * TILE_KT, arow, akc, ncol, bkc, acc);
  }

  // Peeled tail: drain ASYNCcnt fully, no further prefetch.
  asm volatile("s_wait_asynccnt 0x0" ::: "memory");
  __syncthreads();
  mma_step(P.B_bf, &sA[(K_ITERS - 1) & 1][0][0], (K_ITERS - 1) * TILE_KT,
           arow, akc, ncol, bkc, acc);

  // ---- Epilogue: bias + gating + c_t/a_t straight from accumulators. ----
  // C/D layout: element (m,n) lives in VGPR m%8, lane (m/8)*16 + n.
  const int mrow0 = blockM + wm * 64 + ((lane >> 4) << 3);
  const int h     = blockN + wn * 16 + (lane & 15);

  const float bf_ = P.bsum[0 * LSTM_HID + h];
  const float bi_ = P.bsum[1 * LSTM_HID + h];
  const float bo_ = P.bsum[2 * LSTM_HID + h];
  const float bc_ = P.bsum[3 * LSTM_HID + h];

#pragma unroll
  for (int mt = 0; mt < 4; ++mt) {
    const int m0 = mrow0 + mt * 16;
#pragma unroll
    for (int vg = 0; vg < 8; ++vg) {
      const size_t idx = (size_t)(m0 + vg) * LSTM_HID + h;
      const float pf = acc[0][mt][vg] + bf_;
      const float pi = acc[1][mt][vg] + bi_;
      const float po = acc[2][mt][vg] + bo_;
      const float pc = acc[3][mt][vg] + bc_;
      const float fg = sigmoid_f(pf);
      const float ig = sigmoid_f(pi);
      const float og = sigmoid_f(po);
      const float gg = tanhf(pc);
      const float cp = P.c[idx];
      const float ct = fg * cp + ig * gg;
      const float at = og * tanhf(ct);
      P.out_a[idx] = at;
      P.out_c[idx] = ct;
    }
  }
}

extern "C" void kernel_launch(void* const* d_in, const int* in_sizes, int n_in,
                              void* d_out, int out_size, void* d_ws, size_t ws_size,
                              hipStream_t stream) {
  (void)in_sizes; (void)n_in; (void)out_size; (void)ws_size;

  LstmParams P;
  P.x = (const float*)d_in[0];
  P.a = (const float*)d_in[1];
  P.c = (const float*)d_in[2];
  // setup_inputs order: W_f,b_f,U_f,bu_f, W_i,b_i,U_i,bu_i,
  //                     W_o,b_o,U_o,bu_o, W_c,b_c,U_c,bu_c
  P.W[0] = (const float*)d_in[3];  P.b[0]  = (const float*)d_in[4];
  P.U[0] = (const float*)d_in[5];  P.bu[0] = (const float*)d_in[6];
  P.W[1] = (const float*)d_in[7];  P.b[1]  = (const float*)d_in[8];
  P.U[1] = (const float*)d_in[9];  P.bu[1] = (const float*)d_in[10];
  P.W[2] = (const float*)d_in[11]; P.b[2]  = (const float*)d_in[12];
  P.U[2] = (const float*)d_in[13]; P.bu[2] = (const float*)d_in[14];
  P.W[3] = (const float*)d_in[15]; P.b[3]  = (const float*)d_in[16];
  P.U[3] = (const float*)d_in[17]; P.bu[3] = (const float*)d_in[18];

  P.out_a = (float*)d_out;
  P.out_c = (float*)d_out + (size_t)LSTM_BATCH * LSTM_HID;

  // Workspace layout: B_bf (16MB) | A_bf (32MB) | bsum (16KB)
  char* ws = (char*)d_ws;
  P.B_bf = (bf16_t*)ws;
  P.A_bf = (bf16_t*)(ws + (size_t)4 * LSTM_HID * LSTM_KTOT * sizeof(bf16_t));
  P.bsum = (float*)(ws + (size_t)4 * LSTM_HID * LSTM_KTOT * sizeof(bf16_t)
                       + (size_t)LSTM_BATCH * LSTM_KTOT * sizeof(bf16_t));

  // Pass 1: pack weights / activations / bias to bf16 (bandwidth-bound).
  pack_b_kernel<<<dim3((4u * LSTM_HID * LSTM_KTOT) / 8 / 256), 256, 0, stream>>>(P);
  pack_a_kernel<<<dim3(((size_t)LSTM_BATCH * LSTM_KTOT) / 8 / 256), 256, 0, stream>>>(P);
  pack_bias_kernel<<<dim3(16), 256, 0, stream>>>(P);

  // Pass 2: fused WMMA GEMM + gating.
  dim3 grid(LSTM_HID / TILE_BN, LSTM_BATCH / TILE_BM);  // (16, 64)
  lstm_wmma_kernel<<<grid, 256, 0, stream>>>(P);
}
